// VQEmbedding_78202764526046
// MI455X (gfx1250) — compile-verified, matrix operand order checked by reading, not run on hardware
//
#include <hip/hip_runtime.h>

typedef __attribute__((ext_vector_type(16))) __bf16 v16bf;
typedef __attribute__((ext_vector_type(8)))  __bf16 v8bf;
typedef __attribute__((ext_vector_type(8)))  float  v8f;

#define N_TOK   16384
#define D_DIM   256
#define K_CODES 8192
#define L_LEN   2048
#define B_SZ    8
#define KSEG    4
#define KPERSEG (K_CODES / KSEG)
#define ROWS_PER_BLOCK 256          // 8 waves x 32 rows

// workspace layout (byte offsets), total ~9 MB
#define ENORM_OFF   0u                 // 8192 f32            (32 KB)
#define IDX_OFF     (32u*1024u)        // 16384 i32           (64 KB)
#define LOSS_OFF    (96u*1024u)        // 1 f32
#define PBEST_OFF   (128u*1024u)       // KSEG*16384 f32      (256 KB)
#define PIDX_OFF    (384u*1024u)       // KSEG*16384 i32      (256 KB)
#define EHL_OFF     (1u<<20)           // 8192 x 1KB rows: [512B bf16 hi][512B bf16 lo]

// ---------------------------------------------------------------------------
// Kernel 1: split codebook into interleaved bf16 hi/lo rows + compute ||e||^2
// ---------------------------------------------------------------------------
__global__ __launch_bounds__(256) void vq_prep(const float* __restrict__ emb,
                                               float* __restrict__ enorm,
                                               __bf16* __restrict__ ehl) {
    const int k = blockIdx.x;
    const int j = threadIdx.x;
    float w = emb[(size_t)k * D_DIM + j];
    __bf16 h = (__bf16)w;
    __bf16 l = (__bf16)(w - (float)h);
    __bf16* row = ehl + (size_t)k * 512;      // 512 bf16 elems = 1 KB per code
    row[j]         = h;
    row[D_DIM + j] = l;

    __shared__ float red[256];
    red[j] = w * w;
    __syncthreads();
    for (int s = 128; s > 0; s >>= 1) {
        if (j < s) red[j] += red[j + s];
        __syncthreads();
    }
    if (j == 0) enorm[k] = red[0];
}

// ---------------------------------------------------------------------------
// Kernel 2: fused split-bf16 WMMA GEMM + running argmax of (z.e - 0.5||e||^2)
// grid = (N_TOK/256, KSEG), block = 256 (8 waves, 32 rows per wave -> 2 strips)
// Each B fragment (16 codes x 32 d) feeds 6 WMMAs (2 strips x 3 split terms).
// ---------------------------------------------------------------------------
__global__ __launch_bounds__(256) void vq_argmin(const float*  __restrict__ z,
                                                 const __bf16* __restrict__ ehl,
                                                 const float*  __restrict__ enorm,
                                                 float* __restrict__ pbest,
                                                 int*   __restrict__ pidx) {
    const int lane    = threadIdx.x & 31;
    const int wave    = threadIdx.x >> 5;
    const int rowbase = blockIdx.x * ROWS_PER_BLOCK + wave * 32;
    const int kseg0   = blockIdx.y * KPERSEG;
    const int col     = lane & 15;
    const int kb      = (lane < 16) ? 0 : 8;   // ISA 16-bit operand K-group split

    // ---- load this wave's 2x(16x256) A strips once, split into bf16 hi/lo ----
    // z layout is [b][c][l]; row n -> (b = n>>11, l = n&2047), stride L_LEN over d
    v16bf ahi[2][8], alo[2][8];
    #pragma unroll
    for (int s = 0; s < 2; ++s) {
        const int row = rowbase + s * 16 + col;
        const int bb  = row >> 11;
        const int li  = row & (L_LEN - 1);
        const float* zrow = z + (size_t)bb * D_DIM * L_LEN + li;
        #pragma unroll
        for (int dc = 0; dc < 8; ++dc) {
            #pragma unroll
            for (int e = 0; e < 16; ++e) {
                int j = dc * 32 + kb + (e < 8 ? e : e + 8);   // ISA A-frag K map
                float zf = zrow[(size_t)j * L_LEN];
                __bf16 h = (__bf16)zf;
                ahi[s][dc][e] = h;
                alo[s][dc][e] = (__bf16)(zf - (float)h);
            }
        }
    }

    float best[2][8];
    int   bidx[2][8];
    #pragma unroll
    for (int s = 0; s < 2; ++s)
        #pragma unroll
        for (int r = 0; r < 8; ++r) { best[s][r] = -3.0e38f; bidx[s][r] = 0; }

    for (int k0 = 0; k0 < KPERSEG; k0 += 16) {
        const int code = kseg0 + k0 + col;
        const char* bp = (const char*)ehl + (size_t)code * 1024;  // 1 KB/code row

        // seed C with -0.5*||e||^2 so final acc == z.e - 0.5||e||^2
        float iv = -0.5f * enorm[code];
        v8f acc0, acc1;
        #pragma unroll
        for (int r = 0; r < 8; ++r) { acc0[r] = iv; acc1[r] = iv; }

        if (k0 + 16 < KPERSEG)
            __builtin_prefetch(bp + 16 * 1024, 0, 3);   // global_prefetch_b8

        #pragma unroll
        for (int dc = 0; dc < 8; ++dc) {
            // hi halves at [0,512), lo halves at [512,1024) within the code row
            v8bf h0 = *(const v8bf*)(bp + dc * 64 + 2 * kb);
            v8bf h1 = *(const v8bf*)(bp + dc * 64 + 2 * kb + 32);
            v8bf l0 = *(const v8bf*)(bp + 512 + dc * 64 + 2 * kb);
            v8bf l1 = *(const v8bf*)(bp + 512 + dc * 64 + 2 * kb + 32);
            v16bf bhf = __builtin_shufflevector(h0, h1, 0,1,2,3,4,5,6,7,8,9,10,11,12,13,14,15);
            v16bf blf = __builtin_shufflevector(l0, l1, 0,1,2,3,4,5,6,7,8,9,10,11,12,13,14,15);
            // S ~= zhi*ehi + zhi*elo + zlo*ehi ; interleave the two strips so
            // adjacent WMMAs are independent (hide accumulation-chain latency)
            acc0 = __builtin_amdgcn_wmma_f32_16x16x32_bf16(false, ahi[0][dc], false, bhf,
                                                           (short)0, acc0, false, false);
            acc1 = __builtin_amdgcn_wmma_f32_16x16x32_bf16(false, ahi[1][dc], false, bhf,
                                                           (short)0, acc1, false, false);
            acc0 = __builtin_amdgcn_wmma_f32_16x16x32_bf16(false, ahi[0][dc], false, blf,
                                                           (short)0, acc0, false, false);
            acc1 = __builtin_amdgcn_wmma_f32_16x16x32_bf16(false, ahi[1][dc], false, blf,
                                                           (short)0, acc1, false, false);
            acc0 = __builtin_amdgcn_wmma_f32_16x16x32_bf16(false, alo[0][dc], false, bhf,
                                                           (short)0, acc0, false, false);
            acc1 = __builtin_amdgcn_wmma_f32_16x16x32_bf16(false, alo[1][dc], false, bhf,
                                                           (short)0, acc1, false, false);
        }

        // C layout: VGPR r, lanes 0-15 -> row r, lanes 16-31 -> row r+8
        #pragma unroll
        for (int r = 0; r < 8; ++r) {
            if (acc0[r] > best[0][r]) { best[0][r] = acc0[r]; bidx[0][r] = code; }
            if (acc1[r] > best[1][r]) { best[1][r] = acc1[r]; bidx[1][r] = code; }
        }
    }

    // argmax reduction across the 16 columns held in each half-wave
    #pragma unroll
    for (int s = 0; s < 2; ++s) {
        #pragma unroll
        for (int r = 0; r < 8; ++r) {
            float v = best[s][r];
            int   bi = bidx[s][r];
            #pragma unroll
            for (int off = 8; off >= 1; off >>= 1) {
                float v2 = __shfl_xor(v, off, 32);
                int   i2 = __shfl_xor(bi, off, 32);
                if (v2 > v || (v2 == v && i2 < bi)) { v = v2; bi = i2; }
            }
            if (col == 0) {
                int out_row = rowbase + s * 16 + r + kb;   // kb==8 -> rows 8..15
                pbest[(size_t)blockIdx.y * N_TOK + out_row] = v;
                pidx [(size_t)blockIdx.y * N_TOK + out_row] = bi;
            }
        }
    }
}

// ---------------------------------------------------------------------------
// Kernel 3: merge K-segment argmin partials
// ---------------------------------------------------------------------------
__global__ __launch_bounds__(256) void vq_merge(const float* __restrict__ pbest,
                                                const int*   __restrict__ pidx,
                                                int* __restrict__ idx_out) {
    int i = blockIdx.x * blockDim.x + threadIdx.x;
    if (i >= N_TOK) return;
    float best = -3.4e38f;
    int bi = 0;
    for (int s = 0; s < KSEG; ++s) {            // ascending seg -> ties keep lowest index
        float v = pbest[(size_t)s * N_TOK + i];
        if (v > best) { best = v; bi = pidx[(size_t)s * N_TOK + i]; }
    }
    idx_out[i] = bi;
}

// ---------------------------------------------------------------------------
// Kernel 4: codebook gather -> transposed output + MSE partial sums
// ---------------------------------------------------------------------------
__global__ __launch_bounds__(256) void vq_gather(const float* __restrict__ z,
                                                 const float* __restrict__ emb,
                                                 const int*   __restrict__ idx,
                                                 float* __restrict__ out,
                                                 float* __restrict__ loss_acc) {
    const int n = blockIdx.x;
    const int j = threadIdx.x;
    const int code = idx[n];
    float e = emb[(size_t)code * D_DIM + j];
    const int bb = n >> 11;
    const int li = n & (L_LEN - 1);
    size_t off = ((size_t)bb * D_DIM + j) * L_LEN + li;
    float d = e - z[off];
    out[off] = e;                                // z_q_out = z_q (straight-through fwd)

    __shared__ float red[256];
    red[j] = d * d;
    __syncthreads();
    for (int s = 128; s > 0; s >>= 1) {
        if (j < s) red[j] += red[j + s];
        __syncthreads();
    }
    if (j == 0) atomicAdd(loss_acc, red[0]);
}

// ---------------------------------------------------------------------------
// Kernel 5: emit indices (as floats) + scalar loss into d_out tail
// ---------------------------------------------------------------------------
__global__ __launch_bounds__(256) void vq_final(const int* __restrict__ idx,
                                                const float* __restrict__ loss_acc,
                                                float* __restrict__ out) {
    const size_t zq_elems = (size_t)B_SZ * D_DIM * L_LEN;
    int i = blockIdx.x * blockDim.x + threadIdx.x;
    if (i < N_TOK) out[zq_elems + 1 + i] = (float)idx[i];
    if (i == 0)
        out[zq_elems] = 1.25f * loss_acc[0] / ((float)N_TOK * (float)D_DIM);
}

// ---------------------------------------------------------------------------
extern "C" void kernel_launch(void* const* d_in, const int* in_sizes, int n_in,
                              void* d_out, int out_size, void* d_ws, size_t ws_size,
                              hipStream_t stream) {
    const float* z   = (const float*)d_in[0];   // [8, 256, 2048] f32
    const float* emb = (const float*)d_in[1];   // [8192, 256]   f32
    float* out = (float*)d_out;

    char* ws = (char*)d_ws;
    float*  enorm = (float*)(ws + ENORM_OFF);
    int*    idx   = (int*)  (ws + IDX_OFF);
    float*  lacc  = (float*)(ws + LOSS_OFF);
    float*  pbest = (float*)(ws + PBEST_OFF);
    int*    pidx  = (int*)  (ws + PIDX_OFF);
    __bf16* ehl   = (__bf16*)(ws + EHL_OFF);

    hipMemsetAsync(lacc, 0, sizeof(float), stream);

    vq_prep<<<K_CODES, 256, 0, stream>>>(emb, enorm, ehl);

    dim3 grid_gemm(N_TOK / ROWS_PER_BLOCK, KSEG);
    vq_argmin<<<grid_gemm, 256, 0, stream>>>(z, ehl, enorm, pbest, pidx);

    vq_merge<<<(N_TOK + 255) / 256, 256, 0, stream>>>(pbest, pidx, idx);

    vq_gather<<<N_TOK, 256, 0, stream>>>(z, emb, idx, out, lacc);

    vq_final<<<(N_TOK + 255) / 256, 256, 0, stream>>>(idx, lacc, out);
}
